// Mamba2Simple_73701638800218
// MI455X (gfx1250) — matrix operand order repaired and checked by reading, not compiled, not run on hardware
//
#include <hip/hip_runtime.h>

// ---------------- problem dims ----------------
#define BATCH   2
#define LSEQ    4096
#define DMODEL  1024
#define DINNER  2048
#define DSTATE  64
#define HEADDIM 128
#define NHEADS  16
#define DIP     4240   // 2*DINNER + 2*DSTATE + NHEADS
#define CONVD   2176   // DINNER + 2*DSTATE
#define NC      16     // chunks per batch (LSEQ / 256)
#define CHUNK   256
#define ROWS    8192   // BATCH * LSEQ

// ---------------- workspace layout (float elements) ----------------
#define ZX_OFF    0LL
#define CONVX_OFF (ZX_OFF    + (long long)ROWS * DIP)
#define DTB_OFF   (CONVX_OFF + (long long)ROWS * CONVD)
#define ACUM_OFF  (DTB_OFF   + (long long)ROWS * NHEADS)
#define CD_OFF    (ACUM_OFF  + (long long)ROWS * NHEADS)
#define SC_OFF    (CD_OFF    + 512LL)
#define ST_OFF    (SC_OFF    + 32LL * CHUNK * CHUNK)
#define PREV_OFF  (ST_OFF    + 512LL * HEADDIM * DSTATE)
#define Y_OFF     (PREV_OFF  + 512LL * HEADDIM * DSTATE)

typedef __bf16 bf16;
typedef __attribute__((ext_vector_type(16))) __bf16 v16bf;
typedef __attribute__((ext_vector_type(8)))  __bf16 v8bf;
typedef __attribute__((ext_vector_type(8)))  float  v8f;
typedef __attribute__((ext_vector_type(4)))  int    v4i;

// gfx1250 async LDS-copy path (probe via __has_builtin; sync fallback kept)
#if __has_builtin(__builtin_amdgcn_global_load_async_to_lds_b128) && \
    __has_builtin(__builtin_amdgcn_s_wait_asynccnt)
#define HAVE_ASYNC 1
#else
#define HAVE_ASYNC 0
#endif

// ---------------- helpers ----------------
// Split f32 into bf16 hi (truncation) + bf16 lo (residual), pure bit ops.
__device__ __forceinline__ void split_bf(float f, bf16& h, bf16& l) {
  unsigned u  = __float_as_uint(f);
  unsigned hu = u & 0xffff0000u;
  float    lf = f - __uint_as_float(hu);
  unsigned short hs = (unsigned short)(hu >> 16);
  unsigned short ls = (unsigned short)(__float_as_uint(lf) >> 16);
  h = __builtin_bit_cast(bf16, hs);
  l = __builtin_bit_cast(bf16, ls);
}

// bf16x3 fp32-emulating WMMA: acc += (ah+al) x (bh+bl) to 2nd order.
__device__ __forceinline__ v8f wmma3(v16bf ah, v16bf al, v16bf bh, v16bf bl, v8f c) {
  c = __builtin_amdgcn_wmma_f32_16x16x32_bf16(false, ah, false, bl, (short)0, c, false, false);
  c = __builtin_amdgcn_wmma_f32_16x16x32_bf16(false, al, false, bh, (short)0, c, false, false);
  c = __builtin_amdgcn_wmma_f32_16x16x32_bf16(false, ah, false, bh, (short)0, c, false, false);
  return c;
}

// A fragment (16x32 bf16, row-major LDS, row stride ld).
// ISA layout: lanes 0-15 row m=lane hold K {0..7,16..23}; lanes 16-31 hold K {8..15,24..31}.
__device__ __forceinline__ v16bf frag_a_lds(const bf16* lds, int row0, int ld) {
  int lane = threadIdx.x & 31;
  const bf16* p = lds + (row0 + (lane & 15)) * ld + ((lane >> 4) << 3);
  v16bf r;
  *((v8bf*)&r)     = *(const v8bf*)p;
  *((v8bf*)&r + 1) = *(const v8bf*)(p + 16);
  return r;
}

// B fragment from B^T (N x K row-major LDS): lane holds column n=lane%16,
// lanes 0-15 hold K {0..7,16..23}; lanes 16-31 hold K {8..15,24..31}.
__device__ __forceinline__ v16bf frag_b_lds(const bf16* lds, int n0, int ld) {
  int lane = threadIdx.x & 31;
  const bf16* p = lds + (n0 + (lane & 15)) * ld + ((lane >> 4) << 3);
  v16bf r;
  *((v8bf*)&r)     = *(const v8bf*)p;
  *((v8bf*)&r + 1) = *(const v8bf*)(p + 16);
  return r;
}

#if HAVE_ASYNC
// Async DMA: 64 contiguous bytes global -> LDS (4 x b128), no VGPR data path.
// Builtin signature (from clang-22 diagnostic): (v4i AS1* src, v4i AS3* dst, imm off, imm cpol)
__device__ __forceinline__ void async_cp64(const float* g, float* l) {
#pragma unroll
  for (int t = 0; t < 4; t++)
    __builtin_amdgcn_global_load_async_to_lds_b128(
        (__attribute__((address_space(1))) v4i*)(g + 4 * t),
        (__attribute__((address_space(3))) v4i*)(l + 4 * t), 0, 0);
}
#endif

// ---------------- generic bf16x3 GEMM: C = A @ B^T ----------------
// A: M x K (lda), Bt: N x K (ldb), C: M x N (ldc). Block tile 128x128, K step 32.
// Async path: double-buffered raw f32 tiles DMA'd to LDS (ASYNCcnt), converted
// to bf16 hi/lo while the next tile is in flight.
#define GEMM_SMEM (2 * 4096 * 4 * 2 + 4 * 4096 * 2)  // raw A/B x2 bufs + 4 bf16 tiles = 98304 B

__global__ __launch_bounds__(256) void gemm_bf16x3_kernel(
    const float* __restrict__ A, const float* __restrict__ Bt, float* __restrict__ C,
    int M, int N, int K, int lda, int ldb, int ldc,
    long long sA, long long sB, long long sC)
{
  extern __shared__ char gsm[];
  float* Araw = (float*)gsm;         // [2][128*32]
  float* Braw = Araw + 8192;         // [2][128*32]
  bf16*  Ah   = (bf16*)(Braw + 8192);
  bf16*  Al   = Ah + 4096;
  bf16*  Bh   = Al + 4096;
  bf16*  Bl   = Bh + 4096;

  const int z = blockIdx.z;
  A  += (long long)z * sA;
  Bt += (long long)z * sB;
  C  += (long long)z * sC;

  const int m0 = blockIdx.y * 128, n0 = blockIdx.x * 128;
  const int tid = threadIdx.x, lane = tid & 31, wid = tid >> 5;
  const int wm = (wid & 3) * 32;   // wave row offset (4 waves down)
  const int wn = (wid >> 2) * 64;  // wave col offset (2 waves across)

  v8f zero = {};
  v8f acc[2][4];
#pragma unroll
  for (int i = 0; i < 2; i++)
#pragma unroll
    for (int j = 0; j < 4; j++) acc[i][j] = zero;

  const int lr = tid >> 1;          // 0..127 : tile row handled by this thread
  const int lc = (tid & 1) * 16;    // 0 or 16 : 16-float segment
  const bool aval = (m0 + lr) < M;
  const bool bval = (n0 + lr) < N;
  // clamped row addresses (async path copies in-bounds data, zeroed at convert)
  const int arc = aval ? (m0 + lr) : (M - 1);
  const int brc = bval ? (n0 + lr) : (N - 1);
  const long long arow = (long long)arc * lda;
  const long long brow = (long long)brc * ldb;

#if HAVE_ASYNC
  // prologue: DMA first K tile into buffer 0
  async_cp64(A + arow + lc, Araw + lr * 32 + lc);
  async_cp64(Bt + brow + lc, Braw + lr * 32 + lc);

  for (int k0 = 0; k0 < K; k0 += 32) {
    const int cur = (k0 >> 5) & 1;
    float* Ac = Araw + cur * 4096;
    float* Bc = Braw + cur * 4096;
    if (k0 + 32 < K) {  // kick off next tile into the other buffer, then wait for current
      float* An = Araw + (cur ^ 1) * 4096;
      float* Bn = Braw + (cur ^ 1) * 4096;
      async_cp64(A + arow + k0 + 32 + lc, An + lr * 32 + lc);
      async_cp64(Bt + brow + k0 + 32 + lc, Bn + lr * 32 + lc);
      __builtin_amdgcn_s_wait_asynccnt(8);  // 8 per-wave ops of the next tile still in flight
    } else {
      __builtin_amdgcn_s_wait_asynccnt(0);
    }
    __syncthreads();  // all waves' DMA for `cur` visible

    // convert raw f32 tile -> bf16 hi/lo tiles (zero out-of-range rows here)
#pragma unroll
    for (int t = 0; t < 16; t++) {
      bf16 h, l;
      split_bf(aval ? Ac[lr * 32 + lc + t] : 0.f, h, l);
      Ah[lr * 32 + lc + t] = h; Al[lr * 32 + lc + t] = l;
      split_bf(bval ? Bc[lr * 32 + lc + t] : 0.f, h, l);
      Bh[lr * 32 + lc + t] = h; Bl[lr * 32 + lc + t] = l;
    }
    __syncthreads();

    v16bf a_h[2], a_l[2], b_h[4], b_l[4];
#pragma unroll
    for (int i = 0; i < 2; i++) {
      a_h[i] = frag_a_lds(Ah, wm + 16 * i, 32);
      a_l[i] = frag_a_lds(Al, wm + 16 * i, 32);
    }
#pragma unroll
    for (int j = 0; j < 4; j++) {
      b_h[j] = frag_b_lds(Bh, wn + 16 * j, 32);
      b_l[j] = frag_b_lds(Bl, wn + 16 * j, 32);
    }
#pragma unroll
    for (int i = 0; i < 2; i++)
#pragma unroll
      for (int j = 0; j < 4; j++)
        acc[i][j] = wmma3(a_h[i], a_l[i], b_h[j], b_l[j], acc[i][j]);
    __syncthreads();
  }
#else
  // synchronous fallback staging
  for (int k0 = 0; k0 < K; k0 += 32) {
    float fa[16], fb[16];
#pragma unroll
    for (int t = 0; t < 16; t++) { fa[t] = 0.f; fb[t] = 0.f; }
    if (aval) {
      const float4* p = (const float4*)(A + arow + k0 + lc);
#pragma unroll
      for (int t = 0; t < 4; t++) {
        float4 v = p[t];
        fa[4*t+0] = v.x; fa[4*t+1] = v.y; fa[4*t+2] = v.z; fa[4*t+3] = v.w;
      }
    }
    if (bval) {
      const float4* p = (const float4*)(Bt + brow + k0 + lc);
#pragma unroll
      for (int t = 0; t < 4; t++) {
        float4 v = p[t];
        fb[4*t+0] = v.x; fb[4*t+1] = v.y; fb[4*t+2] = v.z; fb[4*t+3] = v.w;
      }
    }
    if (k0 + 32 < K) {
      __builtin_prefetch(A + arow + k0 + 32 + lc, 0, 1);
      __builtin_prefetch(Bt + brow + k0 + 32 + lc, 0, 1);
    }
#pragma unroll
    for (int t = 0; t < 16; t++) {
      bf16 h, l;
      split_bf(fa[t], h, l); Ah[lr * 32 + lc + t] = h; Al[lr * 32 + lc + t] = l;
      split_bf(fb[t], h, l); Bh[lr * 32 + lc + t] = h; Bl[lr * 32 + lc + t] = l;
    }
    __syncthreads();

    v16bf a_h[2], a_l[2], b_h[4], b_l[4];
#pragma unroll
    for (int i = 0; i < 2; i++) {
      a_h[i] = frag_a_lds(Ah, wm + 16 * i, 32);
      a_l[i] = frag_a_lds(Al, wm + 16 * i, 32);
    }
#pragma unroll
    for (int j = 0; j < 4; j++) {
      b_h[j] = frag_b_lds(Bh, wn + 16 * j, 32);
      b_l[j] = frag_b_lds(Bl, wn + 16 * j, 32);
    }
#pragma unroll
    for (int i = 0; i < 2; i++)
#pragma unroll
      for (int j = 0; j < 4; j++)
        acc[i][j] = wmma3(a_h[i], a_l[i], b_h[j], b_l[j], acc[i][j]);
    __syncthreads();
  }
#endif

  // store: lane -> col = n%16, VGPR r -> row r + 8*(lane/16)
#pragma unroll
  for (int i = 0; i < 2; i++)
#pragma unroll
    for (int j = 0; j < 4; j++) {
      union { v8f v; float e[8]; } u; u.v = acc[i][j];
      int col = n0 + wn + 16 * j + (lane & 15);
      if (col < N) {
        int rbase = m0 + wm + 16 * i + ((lane >> 4) << 3);
#pragma unroll
        for (int r = 0; r < 8; r++) {
          int row = rbase + r;
          if (row < M) C[(long long)row * ldc + col] = u.e[r];
        }
      }
    }
}

// ---------------- depthwise causal conv (D_CONV=4) + SiLU ----------------
__global__ __launch_bounds__(256) void conv_silu_kernel(
    const float* __restrict__ zx, const float* __restrict__ cw,
    const float* __restrict__ cb, float* __restrict__ convx)
{
  long long i = (long long)blockIdx.x * 256 + threadIdx.x;
  if (i >= (long long)ROWS * CONVD) return;
  int c = (int)(i % CONVD);
  long long row = i / CONVD;
  int l = (int)(row & (LSEQ - 1));
  long long bbase = row - l;
  float acc = cb[c];
#pragma unroll
  for (int w = 0; w < 4; w++) {
    int ls = l - 3 + w;
    if (ls >= 0) acc += zx[(bbase + ls) * DIP + DINNER + c] * cw[c * 4 + w];
  }
  convx[row * CONVD + c] = acc / (1.f + __expf(-acc));  // silu
}

// ---------------- dt softplus + per-chunk cumsum of dt*A ----------------
__global__ void dt_acum_kernel(const float* __restrict__ zx,
                               const float* __restrict__ dt_bias,
                               const float* __restrict__ A_log,
                               float* __restrict__ dtG, float* __restrict__ acumG,
                               float* __restrict__ cdG)
{
  int bc = blockIdx.x;            // b*NC + c
  int h = threadIdx.x;
  if (h >= NHEADS) return;
  int b = bc >> 4, c = bc & 15;
  float bias = dt_bias[h];
  float Aneg = -__expf(A_log[h]);
  float run = 0.f;
  long long rowbase = (long long)b * LSEQ + c * CHUNK;
  for (int q = 0; q < CHUNK; q++) {
    float x = zx[(rowbase + q) * DIP + (DINNER + CONVD) + h] + bias;
    float dt = (x > 20.f) ? x : log1pf(__expf(x));
    dtG[(rowbase + q) * NHEADS + h] = dt;
    run += dt * Aneg;
    acumG[((long long)bc * CHUNK + q) * NHEADS + h] = run;
  }
  cdG[bc * NHEADS + h] = __expf(run);   // chunk decay, <= 1
}

// ---------------- states[p][n] = sum_q x[q][p] * (dt[q]*exp(acl-ac[q])) * B[q][n] ----------------
__global__ __launch_bounds__(256) void states_kernel(
    const float* __restrict__ convx, const float* __restrict__ acumG,
    const float* __restrict__ dtG, float* __restrict__ statesG)
{
  extern __shared__ char smem[];
  bf16*  xhT = (bf16*)smem;                 // [128][256]  A side: A[p][q]=x[q][p]
  bf16*  xlT = xhT + 128 * 256;
  bf16*  BhT = xlT + 128 * 256;             // [64][256]   B^T[n][q]=w[q]*B[q][n]
  bf16*  BlT = BhT + 64 * 256;
  float* ac_s = (float*)(BlT + 64 * 256);
  float* dt_s = ac_s + 256;

  const int bc = blockIdx.x, h = blockIdx.y;
  const int b = bc >> 4, c = bc & 15;
  const int tid = threadIdx.x, lane = tid & 31, wid = tid >> 5;
  const long long rowbase = (long long)b * LSEQ + c * CHUNK;

  ac_s[tid] = acumG[((long long)bc * CHUNK + tid) * NHEADS + h];
  dt_s[tid] = dtG[(rowbase + tid) * NHEADS + h];
  for (int i = tid; i < 256 * 128; i += 256) {
    int q = i >> 7, p = i & 127;
    float f = convx[(rowbase + q) * CONVD + h * HEADDIM + p];
    bf16 hh, ll; split_bf(f, hh, ll);
    xhT[p * 256 + q] = hh; xlT[p * 256 + q] = ll;
  }
  __syncthreads();
  float acl = ac_s[255];
  for (int i = tid; i < 256 * 64; i += 256) {
    int q = i >> 6, n = i & 63;
    float w = dt_s[q] * __expf(acl - ac_s[q]);
    float f = w * convx[(rowbase + q) * CONVD + DINNER + n];
    bf16 hh, ll; split_bf(f, hh, ll);
    BhT[n * 256 + q] = hh; BlT[n * 256 + q] = ll;
  }
  __syncthreads();

  const int wm = (wid & 3) * 32;   // rows (p): 4 waves * 32 = 128
  const int wn = (wid >> 2) * 32;  // cols (n): 2 waves * 32 = 64
  v8f zero = {};
  v8f acc[2][2];
#pragma unroll
  for (int i = 0; i < 2; i++)
#pragma unroll
    for (int j = 0; j < 2; j++) acc[i][j] = zero;

  for (int kk = 0; kk < 256; kk += 32) {
    v16bf a_h[2], a_l[2], b_h[2], b_l[2];
#pragma unroll
    for (int i = 0; i < 2; i++) {
      a_h[i] = frag_a_lds(xhT + kk, wm + 16 * i, 256);
      a_l[i] = frag_a_lds(xlT + kk, wm + 16 * i, 256);
      b_h[i] = frag_b_lds(BhT + kk, wn + 16 * i, 256);
      b_l[i] = frag_b_lds(BlT + kk, wn + 16 * i, 256);
    }
#pragma unroll
    for (int i = 0; i < 2; i++)
#pragma unroll
      for (int j = 0; j < 2; j++)
        acc[i][j] = wmma3(a_h[i], a_l[i], b_h[j], b_l[j], acc[i][j]);
  }

  float* stg = statesG + ((long long)bc * NHEADS + h) * (HEADDIM * DSTATE);
#pragma unroll
  for (int i = 0; i < 2; i++)
#pragma unroll
    for (int j = 0; j < 2; j++) {
      union { v8f v; float e[8]; } u; u.v = acc[i][j];
      int col = wn + 16 * j + (lane & 15);
      int rbase = wm + 16 * i + ((lane >> 4) << 3);
#pragma unroll
      for (int r = 0; r < 8; r++) stg[(long long)(rbase + r) * DSTATE + col] = u.e[r];
    }
}

// ---------------- sequential inter-chunk scan ----------------
__global__ void scan_kernel(const float* __restrict__ statesG,
                            const float* __restrict__ cdG, float* __restrict__ prevG)
{
  int b = blockIdx.x, h = blockIdx.y;
  for (int i = threadIdx.x; i < HEADDIM * DSTATE; i += 256) {
    float s = 0.f;
    for (int c = 0; c < NC; c++) {
      long long idx = ((long long)(b * NC + c) * NHEADS + h) * (HEADDIM * DSTATE) + i;
      prevG[idx] = s;
      s = cdG[(b * NC + c) * NHEADS + h] * s + statesG[idx];
    }
  }
}

// ---------------- y = M @ x  +  exp(acum)*C @ prev^T  +  Dp * x ----------------
__global__ __launch_bounds__(256) void ydiag_kernel(
    const float* __restrict__ convx, const float* __restrict__ scores,
    const float* __restrict__ acumG, const float* __restrict__ dtG,
    const float* __restrict__ prevG, const float* __restrict__ Dp,
    float* __restrict__ yG)
{
  extern __shared__ char smem[];
  bf16*  xhT = (bf16*)smem;            // [128][256]  B^T[p][k] = x[k][p]
  bf16*  xlT = xhT + 128 * 256;
  float* ac_s = (float*)(xlT + 128 * 256);
  float* dt_s = ac_s + 256;

  const int bc = blockIdx.x, h = blockIdx.y;
  const int b = bc >> 4, c = bc & 15;
  const int tid = threadIdx.x, lane = tid & 31, wid = tid >> 5;
  const long long rowbase = (long long)b * LSEQ + c * CHUNK;

  ac_s[tid] = acumG[((long long)bc * CHUNK + tid) * NHEADS + h];
  dt_s[tid] = dtG[(rowbase + tid) * NHEADS + h];
  for (int i = tid; i < 256 * 128; i += 256) {
    int q = i >> 7, p = i & 127;
    float f = convx[(rowbase + q) * CONVD + h * HEADDIM + p];
    bf16 hh, ll; split_bf(f, hh, ll);
    xhT[p * 256 + q] = hh; xlT[p * 256 + q] = ll;
  }
  __syncthreads();

  const float* sc = scores + (long long)bc * (CHUNK * CHUNK);
  const float* pv = prevG + ((long long)bc * NHEADS + h) * (HEADDIM * DSTATE);
  const float  dp = Dp[h];
  v8f zero = {};

  for (int qt = wid; qt < 16; qt += 8) {   // two 16-row q-tiles per wave
    v8f acc[8];
#pragma unroll
    for (int j = 0; j < 8; j++) acc[j] = zero;

    const int   q   = qt * 16 + (lane & 15);
    const float acq = ac_s[q];
    const int   seg = (lane >> 4) << 3;    // A-frag K segment base (0 or 8)

    // y_diag: A = M (built on the fly), B^T = x^T in LDS, K = 256
    for (int kk = 0; kk < 256; kk += 32) {
      union { v16bf v; bf16 e[16]; } ah, al;
#pragma unroll
      for (int j = 0; j < 16; j++) {
        int k = kk + seg + j + ((j >= 8) ? 8 : 0);
        float m = (k <= q) ? sc[q * 256 + k] * __expf(acq - ac_s[k]) * dt_s[k] : 0.f;
        split_bf(m, ah.e[j], al.e[j]);
      }
#pragma unroll
      for (int pt = 0; pt < 8; pt++) {
        v16bf bh = frag_b_lds(xhT + kk, pt * 16, 256);
        v16bf bl = frag_b_lds(xlT + kk, pt * 16, 256);
        acc[pt] = wmma3(ah.v, al.v, bh, bl, acc[pt]);
      }
    }

    // y_off: A[q][n] = exp(acum_q)*C[q][n], B^T[p][n] = prev[p][n], K = 64
    const float eaq = __expf(acq);
    const float* Cc = convx + rowbase * CONVD + (DINNER + DSTATE);
    for (int kk = 0; kk < 64; kk += 32) {
      union { v16bf v; bf16 e[16]; } ah, al;
#pragma unroll
      for (int j = 0; j < 16; j++) {
        int n = kk + seg + j + ((j >= 8) ? 8 : 0);
        float m = eaq * Cc[(long long)q * CONVD + n];
        split_bf(m, ah.e[j], al.e[j]);
      }
      int ks16 = kk + ((lane >> 4) << 3);
#pragma unroll
      for (int pt = 0; pt < 8; pt++) {
        int p = pt * 16 + (lane & 15);
        union { v16bf v; bf16 e[16]; } bh, bl;
#pragma unroll
        for (int j = 0; j < 16; j++) {
          float f = pv[(long long)p * DSTATE + ks16 + j + ((j >= 8) ? 8 : 0)];
          split_bf(f, bh.e[j], bl.e[j]);
        }
        acc[pt] = wmma3(ah.v, al.v, bh.v, bl.v, acc[pt]);
      }
    }

    // epilogue: + Dp * x, store y
#pragma unroll
    for (int pt = 0; pt < 8; pt++) {
      union { v8f v; float e[8]; } u; u.v = acc[pt];
      int p  = pt * 16 + (lane & 15);
      int rb = qt * 16 + ((lane >> 4) << 3);
#pragma unroll
      for (int r = 0; r < 8; r++) {
        int qq = rb + r;
        float xv = convx[(rowbase + qq) * CONVD + h * HEADDIM + p];
        yG[(rowbase + qq) * DINNER + h * HEADDIM + p] = u.e[r] + dp * xv;
      }
    }
  }
}

// ---------------- gating (silu(z)) + RMSNorm, in place on y ----------------
__global__ __launch_bounds__(256) void gate_norm_kernel(
    const float* __restrict__ zx, const float* __restrict__ nw, float* __restrict__ yG)
{
  __shared__ float red[256];
  long long row = blockIdx.x;
  int tid = threadIdx.x;
  float v[8], s = 0.f;
#pragma unroll
  for (int j = 0; j < 8; j++) {
    int col = tid + j * 256;
    float z = zx[row * DIP + col];
    float g = yG[row * DINNER + col] * (z / (1.f + __expf(-z)));
    v[j] = g; s += g * g;
  }
  red[tid] = s;
  __syncthreads();
  for (int off = 128; off > 0; off >>= 1) {
    if (tid < off) red[tid] += red[tid + off];
    __syncthreads();
  }
  float scale = rsqrtf(red[0] / (float)DINNER + 1e-5f);
#pragma unroll
  for (int j = 0; j < 8; j++) {
    int col = tid + j * 256;
    yG[row * DINNER + col] = v[j] * scale * nw[col];
  }
}

// ---------------- launch ----------------
extern "C" void kernel_launch(void* const* d_in, const int* in_sizes, int n_in,
                              void* d_out, int out_size, void* d_ws, size_t ws_size,
                              hipStream_t stream) {
  (void)in_sizes; (void)n_in; (void)out_size; (void)ws_size;
  const float* u        = (const float*)d_in[0];
  const float* w_in     = (const float*)d_in[1];
  const float* conv_w   = (const float*)d_in[2];
  const float* conv_b   = (const float*)d_in[3];
  const float* dt_bias  = (const float*)d_in[4];
  const float* A_log    = (const float*)d_in[5];
  const float* Dp       = (const float*)d_in[6];
  const float* norm_w   = (const float*)d_in[7];
  const float* w_out    = (const float*)d_in[8];
  float* out = (float*)d_out;
  float* ws  = (float*)d_ws;

  float* zx     = ws + ZX_OFF;
  float* convx  = ws + CONVX_OFF;
  float* dtb    = ws + DTB_OFF;
  float* acum   = ws + ACUM_OFF;
  float* cd     = ws + CD_OFF;
  float* scores = ws + SC_OFF;
  float* states = ws + ST_OFF;
  float* prev   = ws + PREV_OFF;
  float* y      = ws + Y_OFF;

  const int SMEM_ST = (128 * 256 * 2 + 64 * 256 * 2) * (int)sizeof(bf16) + 512 * (int)sizeof(float);
  const int SMEM_YD = (128 * 256 * 2) * (int)sizeof(bf16) + 512 * (int)sizeof(float);
  (void)hipFuncSetAttribute(reinterpret_cast<const void*>(gemm_bf16x3_kernel),
                            hipFuncAttributeMaxDynamicSharedMemorySize, GEMM_SMEM);
  (void)hipFuncSetAttribute(reinterpret_cast<const void*>(states_kernel),
                            hipFuncAttributeMaxDynamicSharedMemorySize, SMEM_ST);
  (void)hipFuncSetAttribute(reinterpret_cast<const void*>(ydiag_kernel),
                            hipFuncAttributeMaxDynamicSharedMemorySize, SMEM_YD);

  // 1) in_proj: zxbct(8192 x 4240) = u(8192 x 1024) @ in_proj_w^T
  gemm_bf16x3_kernel<<<dim3(34, 64, 1), 256, GEMM_SMEM, stream>>>(
      u, w_in, zx, ROWS, DIP, DMODEL, DMODEL, DMODEL, DIP, 0, 0, 0);

  // 2) depthwise conv + SiLU on xBC
  conv_silu_kernel<<<(int)(((long long)ROWS * CONVD) / 256), 256, 0, stream>>>(
      zx, conv_w, conv_b, convx);

  // 3) dt softplus, per-chunk cumsum, chunk decay
  dt_acum_kernel<<<32, 64, 0, stream>>>(zx, dt_bias, A_log, dtb, acum, cd);

  // 4) scores = C_chunk @ B_chunk^T (batched over 32 (b,c) chunks)
  gemm_bf16x3_kernel<<<dim3(2, 2, 32), 256, GEMM_SMEM, stream>>>(
      convx + (DINNER + DSTATE), convx + DINNER, scores,
      CHUNK, CHUNK, DSTATE, CONVD, CONVD, CHUNK,
      (long long)CHUNK * CONVD, (long long)CHUNK * CONVD, (long long)CHUNK * CHUNK);

  // 5) per-(b,c,h) chunk states
  states_kernel<<<dim3(32, NHEADS), 256, SMEM_ST, stream>>>(convx, acum, dtb, states);

  // 6) inter-chunk exclusive scan
  scan_kernel<<<dim3(BATCH, NHEADS), 256, 0, stream>>>(states, cd, prev);

  // 7) y_diag + y_off + skip
  ydiag_kernel<<<dim3(32, NHEADS), 256, SMEM_YD, stream>>>(
      convx, scores, acum, dtb, prev, Dp, y);

  // 8) gate + RMSNorm (in place)
  gate_norm_kernel<<<ROWS, 256, 0, stream>>>(zx, norm_w, y);

  // 9) out_proj: out(8192 x 1024) = y @ out_proj_w^T
  gemm_bf16x3_kernel<<<dim3(8, 64, 1), 256, GEMM_SMEM, stream>>>(
      y, w_out, out, ROWS, DMODEL, DINNER, DINNER, DINNER, DMODEL, 0, 0, 0);
}
